// Transformer_22187801051725
// MI455X (gfx1250) — matrix operand order repaired and checked by reading, not compiled, not run on hardware
//
#include <hip/hip_runtime.h>
#include <cstdint>

// ---------------------------------------------------------------------------
// Transformer forward for MI455X (gfx1250, wave32, WMMA, async-to-LDS).
//
// All GEMMs: bf16 operands, v_wmma_f32_16x16x32_bf16 with fp32 accumulate.
// Weights are pre-converted fp32 -> bf16 AND pre-transposed to N-major, so
// both GEMM operands are K-contiguous; tiles are staged into double-buffered
// LDS with per-lane global_load_async_to_lds_b128 (ASYNCcnt), prefetching
// tile i+1 while tile i is consumed by the WMMAs. No bounds guards needed:
// all M mult of 128, K mult of 32, N mult of 128 (except P@V with N=64,
// which uses a small guarded synchronous path for B, template BT=0).
// Producers (embed, add+LN, softmax, GEMM epilogue) emit bf16 mirrors.
// ---------------------------------------------------------------------------

#define BM 128
#define BN 128
#define BKD 32
#define NTHREADS 256

typedef __attribute__((ext_vector_type(16))) __bf16 v16bf;
typedef __attribute__((ext_vector_type(8)))  float  v8f;
typedef unsigned short ushort_t;

__device__ __forceinline__ ushort_t f2bf(float f) {
  union { float f; unsigned int u; } x; x.f = f;
  unsigned int r = x.u + 0x7FFFu + ((x.u >> 16) & 1u);   // round-to-nearest-even
  return (ushort_t)(r >> 16);
}

union FragU {
  v16bf v;
  uint4 q[2];
};

union Pack8 {
  uint4 q;
  ushort_t s[8];
};

// Stage a [128 rows][32 k] bf16 tile from a row-major (K-contiguous) source
// into LDS using async 16B copies. 512 chunks, 2 per thread (2 async
// instructions per wave). No guards: caller guarantees rows and K in range.
__device__ __forceinline__ void stage_tile_async(ushort_t (*ls)[BKD],
                                                 const ushort_t* __restrict__ src,
                                                 int rowStride, int row0, int k0,
                                                 int tid)
{
#pragma unroll
  for (int it = 0; it < 2; ++it) {
    const int c   = tid + it * 256;      // 0..511
    const int row = c >> 2;              // 0..127
    const int kc  = (c & 3) << 3;        // 0,8,16,24
    unsigned lds_off = (unsigned)(size_t)&ls[row][kc];
    const ushort_t* gp = src + (size_t)(row0 + row) * rowStride + (k0 + kc);
    asm volatile("global_load_async_to_lds_b128 %0, %1, off"
                 :: "v"(lds_off), "v"(gp) : "memory");
  }
}

// C[M,N] = act( A[M,K] @ B + bias ); A,B bf16; C fp32 (optional) and/or bf16
// mirror Cbf (optional). Batched over blockIdx.z with split strides.
// BT==1: B is N-major (B[n*ldb+k]) -> async path (4 async ops/wave/tile).
// BT==0: B is K-major (B[k*ldb+n]) -> guarded sync scatter (P@V only, N=64;
//        2 async ops/wave/tile for A).
template <int BT>
__global__ __launch_bounds__(NTHREADS)
void gemm_bf16_wmma(const ushort_t* __restrict__ A, const ushort_t* __restrict__ B,
                    const float* __restrict__ bias,
                    float* __restrict__ C, ushort_t* __restrict__ Cbf,
                    int M, int N, int K, int lda, int ldb, int ldc,
                    int act, int Hdiv,
                    long long sA1, long long sA2,
                    long long sB1, long long sB2,
                    long long sC1, long long sC2)
{
  __shared__ __align__(16) ushort_t lsA[2][BM][BKD];  // [stage][m][k]
  __shared__ __align__(16) ushort_t lsB[2][BN][BKD];  // [stage][n][k]
  (void)M;

  const int tid = threadIdx.x;
  const int z   = blockIdx.z;
  const int zo  = z / Hdiv;
  const int zi  = z % Hdiv;
  const ushort_t* Ab = A + (size_t)((long long)zo * sA1 + (long long)zi * sA2);
  const ushort_t* Bb = B + (size_t)((long long)zo * sB1 + (long long)zi * sB2);
  const long long cOff = (long long)zo * sC1 + (long long)zi * sC2;

  const int tileM0 = blockIdx.y * BM;
  const int tileN0 = blockIdx.x * BN;

  const int lane = tid & 31;
  const int wave = tid >> 5;              // 8 waves
  const int wm   = (wave >> 1) * 32;      // 0,32,64,96
  const int wn   = (wave & 1) * 64;       // 0,64
  const int half = lane >> 4;
  const int l15  = lane & 15;

  v8f acc[2][4];
  const v8f vzero = {0.f,0.f,0.f,0.f,0.f,0.f,0.f,0.f};
#pragma unroll
  for (int mi = 0; mi < 2; ++mi)
#pragma unroll
    for (int ni = 0; ni < 4; ++ni) acc[mi][ni] = vzero;

  // Stage both tiles for step `i` into LDS buffer `buf`.
  auto stage = [&](int buf, int k0) {
    stage_tile_async(lsA[buf], Ab, lda, tileM0, k0, tid);
    if (BT) {
      stage_tile_async(lsB[buf], Bb, ldb, tileN0, k0, tid);
    } else {
      // lsB[buf][n][k] = B[(k0+k)*ldb + tileN0+n]
      const int n8 = (tid & 15) << 3;   // 0..120
      const int kk = tid >> 4;          // 0..15
#pragma unroll
      for (int r = 0; r < 2; ++r) {
        const int k  = kk + 16 * r;
        const int gk = k0 + k;
        Pack8 w;
        if (tileN0 + n8 + 8 <= N) {
          w.q = *(const uint4*)(Bb + (size_t)gk * ldb + tileN0 + n8);
        } else {
          w.q = make_uint4(0u, 0u, 0u, 0u);
        }
#pragma unroll
        for (int j = 0; j < 8; ++j) lsB[buf][n8 + j][k] = w.s[j];
      }
    }
  };

  const int nk = K / BKD;
  stage(0, 0);
  for (int i = 0; i < nk; ++i) {
    const int cur = i & 1;
    const bool more = (i + 1 < nk);
    if (more) stage(cur ^ 1, (i + 1) * BKD);   // prefetch next tile

    // Wait only for the current tile's async copies; leave the next tile's
    // (4 for BT=1, 2 for BT=0) in flight. Async ops complete in order.
    if (more) {
      if (BT) asm volatile("s_wait_asynccnt 0x4" ::: "memory");
      else    asm volatile("s_wait_asynccnt 0x2" ::: "memory");
    } else {
      asm volatile("s_wait_asynccnt 0x0" ::: "memory");
    }
    __syncthreads();

    // ---- fragments per ISA 7.12.2 (16-bit A 16x32; B 32x16) ----
    FragU fa[2], fb[4];
#pragma unroll
    for (int mi = 0; mi < 2; ++mi) {
      const int row = wm + mi * 16 + l15;
      fa[mi].q[0] = *(const uint4*)&lsA[cur][row][half * 8];       // K = 8h..8h+7
      fa[mi].q[1] = *(const uint4*)&lsA[cur][row][16 + half * 8];  // K = 16+8h..
    }
#pragma unroll
    for (int ni = 0; ni < 4; ++ni) {
      const int row = wn + ni * 16 + l15;
      fb[ni].q[0] = *(const uint4*)&lsB[cur][row][half * 8];       // K = 8h..8h+7
      fb[ni].q[1] = *(const uint4*)&lsB[cur][row][16 + half * 8];  // K = 16+8h..
    }

#pragma unroll
    for (int mi = 0; mi < 2; ++mi)
#pragma unroll
      for (int ni = 0; ni < 4; ++ni)
        acc[mi][ni] = __builtin_amdgcn_wmma_f32_16x16x32_bf16(
            false, fa[mi].v, false, fb[ni].v, (short)0, acc[mi][ni], false, false);

    // All waves must finish reading buf `cur` before step i+1 stages into it.
    // (ds_loads are complete here: the WMMAs above forced s_wait_dscnt 0.)
    __syncthreads();
  }

  // ---- epilogue (M mult of 128 -> no row guard; guard only n) ----
  float* Cb = C ? C + (size_t)cOff : nullptr;
  ushort_t* Cm = Cbf ? Cbf + (size_t)cOff : nullptr;
#pragma unroll
  for (int mi = 0; mi < 2; ++mi) {
#pragma unroll
    for (int ni = 0; ni < 4; ++ni) {
      const int gn = tileN0 + wn + ni * 16 + l15;
      if (gn >= N) continue;
      const float bv = bias ? bias[gn] : 0.0f;
#pragma unroll
      for (int r = 0; r < 8; ++r) {
        const int gm = tileM0 + wm + mi * 16 + r + half * 8;
        float v = acc[mi][ni][r] + bv;
        if (act == 1) v = fmaxf(v, 0.0f);
        const size_t idx = (size_t)gm * ldc + gn;
        if (Cb) Cb[idx] = v;
        if (Cm) Cm[idx] = f2bf(v);
      }
    }
  }
}

// Transpose + convert: out[n*K + k] = bf16(in[k*N + n]); in is K x N fp32.
__global__ void convT_kernel(const float* __restrict__ in, ushort_t* __restrict__ out,
                             int K, int N)
{
  __shared__ float t[32][33];
  const int kb = blockIdx.y * 32;
  const int nb = blockIdx.x * 32;
  const int tx = threadIdx.x & 31;
  const int ty = threadIdx.x >> 5;    // 0..7
#pragma unroll
  for (int r = ty; r < 32; r += 8) {
    const int k = kb + r, n = nb + tx;
    t[r][tx] = (k < K && n < N) ? in[(size_t)k * N + n] : 0.f;
  }
  __syncthreads();
#pragma unroll
  for (int r = ty; r < 32; r += 8) {
    const int n = nb + r, k = kb + tx;
    if (n < N && k < K) out[(size_t)n * K + k] = f2bf(t[tx][r]);
  }
}

// out fp32 + bf16 mirror: emb[tok][d] + sinusoidal PE(s, d)
__global__ void embed_pe_kernel(const int* __restrict__ tok,
                                const float* __restrict__ emb,
                                float* __restrict__ out, ushort_t* __restrict__ outbf,
                                int Seq, int Dm)
{
  const int i = blockIdx.x * blockDim.x + threadIdx.x;
  const int d = i % Dm;
  const int row = i / Dm;
  const int s = row % Seq;
  const int t = tok[row];
  const int hd = d >> 1;
  const float dv = expf(-(float)(2 * hd) * logf(10000.0f) / (float)Dm);
  const float ang = (float)s * dv;
  const float pe = (d & 1) ? cosf(ang) : sinf(ang);
  const float v = emb[(size_t)t * Dm + d] + pe;
  out[i] = v;
  outbf[i] = f2bf(v);
}

// In-place masked, scaled softmax; also writes bf16 mirror.
__global__ void softmax_kernel(float* __restrict__ S, ushort_t* __restrict__ Sbf,
                               int cols, int rowsPer, int causal, float scale)
{
  const int row = blockIdx.x;
  const int i = row % rowsPer;
  float* p = S + (size_t)row * cols;
  ushort_t* pb = Sbf + (size_t)row * cols;
  __shared__ float red[256];
  const int tid = threadIdx.x;

  float mx = -3.4e38f;
  for (int j = tid; j < cols; j += 256) {
    float v = p[j] * scale;
    if (causal && j > i) v = -1e30f;
    p[j] = v;
    mx = fmaxf(mx, v);
  }
  red[tid] = mx; __syncthreads();
  for (int s2 = 128; s2 > 0; s2 >>= 1) {
    if (tid < s2) red[tid] = fmaxf(red[tid], red[tid + s2]);
    __syncthreads();
  }
  mx = red[0]; __syncthreads();

  float sum = 0.f;
  for (int j = tid; j < cols; j += 256) {
    float e = expf(p[j] - mx);
    p[j] = e;
    sum += e;
  }
  red[tid] = sum; __syncthreads();
  for (int s2 = 128; s2 > 0; s2 >>= 1) {
    if (tid < s2) red[tid] += red[tid + s2];
    __syncthreads();
  }
  const float inv = 1.0f / red[0];
  for (int j = tid; j < cols; j += 256) {
    float v = p[j] * inv;
    p[j] = v;
    pb[j] = f2bf(v);
  }
}

// y = LayerNorm(x + a)*g + b ; writes fp32 y (may alias x) + bf16 mirror.
__global__ void add_ln_kernel(const float* __restrict__ x, const float* __restrict__ a,
                              const float* __restrict__ g, const float* __restrict__ b,
                              float* __restrict__ y, ushort_t* __restrict__ ybf, int Dm)
{
  const int row = blockIdx.x;
  const int tid = threadIdx.x;
  const float* xr = x + (size_t)row * Dm;
  const float* ar = a + (size_t)row * Dm;
  float* yr = y + (size_t)row * Dm;
  ushort_t* yb = ybf + (size_t)row * Dm;
  __shared__ float r1[256], r2[256];
  float s = 0.f, s2 = 0.f;
  for (int d = tid; d < Dm; d += 256) {
    float v = xr[d] + ar[d];
    s += v; s2 += v * v;
  }
  r1[tid] = s; r2[tid] = s2; __syncthreads();
  for (int k = 128; k > 0; k >>= 1) {
    if (tid < k) { r1[tid] += r1[tid + k]; r2[tid] += r2[tid + k]; }
    __syncthreads();
  }
  const float mean = r1[0] / (float)Dm;
  const float var  = r2[0] / (float)Dm - mean * mean;
  const float inv  = rsqrtf(var + 1e-5f);
  for (int d = tid; d < Dm; d += 256) {
    float v = (xr[d] + ar[d] - mean) * inv * g[d] + b[d];
    yr[d] = v;
    yb[d] = f2bf(v);
  }
}

// ---------------------------------------------------------------------------
// Input order assumption (top-level dict insertion order; params flattened as
// JAX tree leaves, alphabetical recursive):
//  0 src  1 tgt  2 tgt_mask
//  3 Wout 4 bout
//  5 dec.W1 6 dec.W2 7 dec.b1 8 dec.b2 9 dec.be1 10 dec.be2 11 dec.be3
//  12..19 dec.ca: Wk,Wo,Wq,Wv,bk,bo,bq,bv
//  20 dec.g1 21 dec.g2 22 dec.g3
//  23..30 dec.sa: Wk,Wo,Wq,Wv,bk,bo,bq,bv
//  31 dec_emb
//  32 enc.W1 33 enc.W2 34 enc.b1 35 enc.b2 36 enc.be1 37 enc.be2 38 enc.g1 39 enc.g2
//  40..47 enc.sa: Wk,Wo,Wq,Wv,bk,bo,bq,bv
//  48 enc_emb
// ---------------------------------------------------------------------------
extern "C" void kernel_launch(void* const* d_in, const int* in_sizes, int n_in,
                              void* d_out, int out_size, void* d_ws, size_t ws_size,
                              hipStream_t stream)
{
  (void)in_sizes; (void)out_size; (void)ws_size;
  if (n_in < 49) return;

  const int Bsz = 2, Sq = 1024, Dm = 768, Hh = 12, Ff = 3072, Ll = 4, Vv = 32000, HDm = 64;
  const int BS = Bsz * Sq;  // 2048 rows

  auto F = [&](int i) { return (const float*)d_in[i]; };
  const int* src = (const int*)d_in[0];
  const int* tgt = (const int*)d_in[1];

  // ---- workspace carve-out (~245 MB) ----
  char* wsB = (char*)d_ws;
  auto allocF = [&](size_t n) { float* p = (float*)wsB; wsB += ((n * 4 + 255) & ~(size_t)255); return p; };
  auto allocH = [&](size_t n) { ushort_t* p = (ushort_t*)wsB; wsB += ((n * 2 + 255) & ~(size_t)255); return p; };

  float*    bufX  = allocF((size_t)BS * Dm);            // fp32 residual stream
  float*    bufT  = allocF((size_t)BS * Dm);            // fp32 sublayer output
  float*    bufS  = allocF((size_t)Bsz * Hh * Sq * Sq); // fp32 scores/probs
  ushort_t* xbf   = allocH((size_t)BS * Dm);            // bf16 mirror of bufX
  ushort_t* encbf = allocH((size_t)BS * Dm);            // bf16 encoder output
  ushort_t* qbf   = allocH((size_t)BS * Dm);
  ushort_t* kbf   = allocH((size_t)BS * Dm);
  ushort_t* vbf   = allocH((size_t)BS * Dm);
  ushort_t* abf   = allocH((size_t)BS * Dm);            // attention concat
  ushort_t* hbf   = allocH((size_t)BS * Ff);            // FFN hidden
  ushort_t* sbf   = allocH((size_t)Bsz * Hh * Sq * Sq); // bf16 probs
  ushort_t* wbf   = allocH((size_t)Dm * Vv);            // transposed bf16 weight scratch

  auto gemm = [&](const ushort_t* A, const ushort_t* Bm, const float* bias,
                  float* C, ushort_t* Cbf,
                  int M, int N, int K, int lda, int ldb, int ldc, int BT, int act,
                  int batch, int Hdiv,
                  long long sA1, long long sA2, long long sB1, long long sB2,
                  long long sC1, long long sC2) {
    dim3 grid((N + BN - 1) / BN, (M + BM - 1) / BM, batch);
    if (BT)
      gemm_bf16_wmma<1><<<grid, NTHREADS, 0, stream>>>(
          A, Bm, bias, C, Cbf, M, N, K, lda, ldb, ldc, act, Hdiv,
          sA1, sA2, sB1, sB2, sC1, sC2);
    else
      gemm_bf16_wmma<0><<<grid, NTHREADS, 0, stream>>>(
          A, Bm, bias, C, Cbf, M, N, K, lda, ldb, ldc, act, Hdiv,
          sA1, sA2, sB1, sB2, sC1, sC2);
  };

  // Convert+transpose a KxN fp32 weight into wbf (N-major bf16).
  auto convT = [&](const float* W, int K, int N) {
    dim3 grid((N + 31) / 32, (K + 31) / 32);
    convT_kernel<<<grid, 256, 0, stream>>>(W, wbf, K, N);
  };

  // Dense projection: out = act(in @ W + b), W fp32 KxN (layer l slice).
  auto proj = [&](const ushort_t* in, const float* W, const float* b, int l,
                  int K, int N, float* outF, ushort_t* outBf, int act) {
    convT(W + (size_t)l * K * N, K, N);
    gemm(in, wbf, b ? b + (size_t)l * N : nullptr, outF, outBf,
         BS, N, K, K, K, N, /*BT=*/1, act, 1, 1, 0,0,0,0,0,0);
  };

  // Full multi-head attention -> bufT (pre-residual).
  auto attn = [&](const ushort_t* xq, const ushort_t* xkv, int l,
                  const float* Wq, const float* bq, const float* Wk, const float* bk,
                  const float* Wv, const float* bv, const float* Wo, const float* bo,
                  int causal) {
    proj(xq,  Wq, bq, l, Dm, Dm, nullptr, qbf, 0);
    proj(xkv, Wk, bk, l, Dm, Dm, nullptr, kbf, 0);
    proj(xkv, Wv, bv, l, Dm, Dm, nullptr, vbf, 0);
    // scores[b,h] = Q[b,:,h*64:] @ K[b,:,h*64:]^T  (24 batches)
    gemm(qbf, kbf, nullptr, bufS, nullptr, Sq, Sq, HDm, Dm, Dm, Sq, /*BT=*/1, 0,
         Bsz * Hh, Hh,
         (long long)Sq * Dm, (long long)HDm,
         (long long)Sq * Dm, (long long)HDm,
         (long long)Hh * Sq * Sq, (long long)Sq * Sq);
    softmax_kernel<<<Bsz * Hh * Sq, 256, 0, stream>>>(bufS, sbf, Sq, Sq, causal, 0.125f);
    // attn_out[b,:,h*64:] = P[b,h] @ V[b,:,h*64:]   (B is K-major, N=64)
    gemm(sbf, vbf, nullptr, nullptr, abf, Sq, HDm, Sq, Sq, Dm, Dm, /*BT=*/0, 0,
         Bsz * Hh, Hh,
         (long long)Hh * Sq * Sq, (long long)Sq * Sq,
         (long long)Sq * Dm, (long long)HDm,
         (long long)Sq * Dm, (long long)HDm);
    proj(abf, Wo, bo, l, Dm, Dm, bufT, nullptr, 0);
  };

  auto ffn = [&](const float* W1, const float* b1, const float* W2, const float* b2, int l) {
    proj(xbf, W1, b1, l, Dm, Ff, nullptr, hbf, /*relu*/1);
    proj(hbf, W2, b2, l, Ff, Dm, bufT, nullptr, 0);
  };

  auto add_ln = [&](const float* g, const float* b, int l) {
    add_ln_kernel<<<BS, 256, 0, stream>>>(bufX, bufT, g + (size_t)l * Dm,
                                          b + (size_t)l * Dm, bufX, xbf, Dm);
  };

  // ---------------- Encoder ----------------
  embed_pe_kernel<<<(BS * Dm) / 256, 256, 0, stream>>>(src, F(48), bufX, xbf, Sq, Dm);
  for (int l = 0; l < Ll; ++l) {
    attn(xbf, xbf, l, F(42), F(46), F(40), F(44), F(43), F(47), F(41), F(45), /*causal*/0);
    add_ln(F(38), F(36), l);                              // g1, be1
    ffn(F(32), F(34), F(33), F(35), l);                   // W1,b1,W2,b2
    add_ln(F(39), F(37), l);                              // g2, be2
  }
  hipMemcpyAsync(encbf, xbf, (size_t)BS * Dm * sizeof(ushort_t),
                 hipMemcpyDeviceToDevice, stream);

  // ---------------- Decoder ----------------
  embed_pe_kernel<<<(BS * Dm) / 256, 256, 0, stream>>>(tgt, F(31), bufX, xbf, Sq, Dm);
  for (int l = 0; l < Ll; ++l) {
    attn(xbf, xbf, l, F(25), F(29), F(23), F(27), F(26), F(30), F(24), F(28), /*causal*/1);
    add_ln(F(20), F(9), l);                               // g1, be1
    attn(xbf, encbf, l, F(14), F(18), F(12), F(16), F(15), F(19), F(13), F(17), /*causal*/0);
    add_ln(F(21), F(10), l);                              // g2, be2
    ffn(F(5), F(7), F(6), F(8), l);                       // W1,b1,W2,b2
    add_ln(F(22), F(11), l);                              // g3, be3
  }

  // ---------------- Output projection ----------------
  convT(F(3), Dm, Vv);
  gemm(xbf, wbf, F(4), (float*)d_out, nullptr, BS, Vv, Dm, Dm, Dm, Vv, /*BT=*/1, 0,
       1, 1, 0,0,0,0,0,0);
}
